// PointwiseSoftMask_27762668601764
// MI455X (gfx1250) — matrix operand order repaired and checked by reading, not compiled, not run on hardware
//
#include <hip/hip_runtime.h>
#include <math.h>

// ---------- types ----------
typedef __attribute__((ext_vector_type(16))) __bf16 v16bf;
typedef __attribute__((ext_vector_type(8)))  float  v8f;
typedef unsigned short u16;
typedef unsigned int   u32;
typedef unsigned long long u64;

union Frag { uint4 q[2]; v16bf v; };   // 32 bytes = 16 bf16 = one WMMA A/B operand

#define N_ROWS 4096
#define M_COLS 8192
#define DDIM   512
#define EPSF   1e-12f

// async global->LDS DMA (CDNA5): LDS addr VGPR, 64-bit global addr VGPR pair
#define ASYNC_LD_B128(ldsaddr, gaddr) \
    asm volatile("global_load_async_to_lds_b128 %0, %1, off" \
                 :: "v"(ldsaddr), "v"(gaddr) : "memory")
#define WAIT_ASYNC_8() asm volatile("s_wait_asynccnt 0x8" ::: "memory")
#define WAIT_ASYNC_0() asm volatile("s_wait_asynccnt 0x0" ::: "memory")

// monotone float -> uint key (order-preserving), for integer atomic min/max
__device__ __forceinline__ u32 fkey(float f) {
    u32 u = __float_as_uint(f);
    return (u & 0x80000000u) ? ~u : (u | 0x80000000u);
}
__device__ __forceinline__ float fdec(u32 k) {
    return __uint_as_float((k & 0x80000000u) ? (k & 0x7fffffffu) : ~k);
}

__device__ __forceinline__ u16 f32_to_bf16_rne(float f) {
    u32 u = __float_as_uint(f);
    u32 r = u + 0x7fffu + ((u >> 16) & 1u);
    return (u16)(r >> 16);
}
__device__ __forceinline__ float bf16_to_f32(u16 h) {
    return __uint_as_float(((u32)h) << 16);
}

// ---------- kernel 1: split f32 -> bf16 hi/lo copies (x and x_n) ----------
// grid = 12288 blocks (rows of x then x_n), 128 threads, one float4 each
__global__ __launch_bounds__(128)
void split_kernel(const float* __restrict__ x, const float* __restrict__ xn,
                  u16* __restrict__ xhi, u16* __restrict__ xlo,
                  u16* __restrict__ nhi, u16* __restrict__ nlo) {
    int row = blockIdx.x;
    int t   = threadIdx.x;
    const float* src; u16* dhi; u16* dlo; size_t r;
    if (row < N_ROWS) { src = x;  dhi = xhi; dlo = xlo; r = row; }
    else              { src = xn; dhi = nhi; dlo = nlo; r = row - N_ROWS; }
    size_t base = r * DDIM + (size_t)t * 4;
    float4 v = *(const float4*)(src + base);
    float f[4] = { v.x, v.y, v.z, v.w };
    u16 h2[4], l2[4];
#pragma unroll
    for (int e = 0; e < 4; ++e) {
        u16 h = f32_to_bf16_rne(f[e]);
        float rem = f[e] - bf16_to_f32(h);
        h2[e] = h;
        l2[e] = f32_to_bf16_rne(rem);
    }
    *(uint2*)(dhi + base) = make_uint2((u32)h2[0] | ((u32)h2[1] << 16),
                                       (u32)h2[2] | ((u32)h2[3] << 16));
    *(uint2*)(dlo + base) = make_uint2((u32)l2[0] | ((u32)l2[1] << 16),
                                       (u32)l2[2] | ((u32)l2[3] << 16));
}

// ---------- kernel 2: per-row gate params + stat-key init ----------
__global__ __launch_bounds__(256)
void rowp_kernel(const float* __restrict__ x, const float* __restrict__ W,
                 const float* __restrict__ bias,
                 float* __restrict__ rowA, float* __restrict__ rowB,
                 float* __restrict__ rowIS,
                 u32* __restrict__ minK, u32* __restrict__ maxK) {
    int row = blockIdx.x * blockDim.x + threadIdx.x;
    if (row >= N_ROWS) return;
    const float4* xr = (const float4*)(x + (size_t)row * DDIM);
    const float4* w0 = (const float4*)(W);
    const float4* w1 = (const float4*)(W + DDIM);
    const float4* w2 = (const float4*)(W + 2 * DDIM);
    float a0 = 0.f, a1 = 0.f, a2 = 0.f;
    for (int i = 0; i < DDIM / 4; ++i) {
        float4 xv = xr[i], u = w0[i], v = w1[i], s = w2[i];
        a0 += xv.x * u.x + xv.y * u.y + xv.z * u.z + xv.w * u.w;
        a1 += xv.x * v.x + xv.y * v.y + xv.z * v.z + xv.w * v.w;
        a2 += xv.x * s.x + xv.y * s.y + xv.z * s.z + xv.w * s.w;
    }
    a0 += bias[0]; a1 += bias[1]; a2 += bias[2];
    float pa = 1.0f / (1.0f + expf(-a0));
    float pb = 1.0f / (1.0f + expf(-a1));
    float ps = 1.0f / (1.0f + expf(-a2));
    pa = fminf(fmaxf(pa, EPSF), 1.0f - EPSF);
    pb = fminf(fmaxf(pb, EPSF), 1.0f - EPSF);
    ps = fminf(fmaxf(ps, EPSF), 1.0f - EPSF);
    float bb = pa + pb * (1.0f - pa);     // b = a + b*(1-a)
    rowA[row]  = pa;
    rowB[row]  = bb;
    rowIS[row] = 1.0f / ps;
    minK[row]  = 0xFFFFFFFFu;   // +inf key
    maxK[row]  = 0u;            // -inf key
}

// ---------- kernel 3: WMMA GEMM, double-buffered async global->LDS ----------
// sim = x @ x_n^T / D ; per-row min/max folded in via keyed atomics.
// grid = (M/128, N/128), 256 threads (8 waves); wave w owns rows w*16..w*16+15
__global__ __launch_bounds__(256)
void gemm_wmma(const u16* __restrict__ xhi, const u16* __restrict__ xlo,
               const u16* __restrict__ nhi, const u16* __restrict__ nlo,
               float* __restrict__ out,
               u32* __restrict__ minK, u32* __restrict__ maxK) {
    // [buffer][matrix: 0=Ahi 1=Alo 2=Bhi 3=Blo][128 rows x 32 k]  -> 64 KB
    __shared__ alignas(16) u16 lds[2][4][128 * 32];

    const int t    = threadIdx.x;
    const int lane = t & 31;
    const int w    = t >> 5;
    const int row0 = blockIdx.y * 128;
    const int col0 = blockIdx.x * 128;

    // per-thread cooperative-copy coordinates (2 b128 chunks per matrix)
    const int r0i = t >> 2,          c0i = (t & 3) << 3;           // rows 0..63
    const int r1i = (t + 256) >> 2,  c1i = ((t + 256) & 3) << 3;   // rows 64..127

    const u16* gsrc[4] = { xhi + (size_t)row0 * DDIM, xlo + (size_t)row0 * DDIM,
                           nhi + (size_t)col0 * DDIM, nlo + (size_t)col0 * DDIM };

    // issue 8 async b128 DMAs (this thread's share of one 32KB k-tile stage)
    auto issue = [&](int buf, int k0) {
#pragma unroll
        for (int m = 0; m < 4; ++m) {
            u32 l0 = (u32)(uintptr_t)&lds[buf][m][r0i * 32 + c0i];
            u32 l1 = (u32)(uintptr_t)&lds[buf][m][r1i * 32 + c1i];
            u64 g0 = (u64)(uintptr_t)(gsrc[m] + (size_t)r0i * DDIM + k0 + c0i);
            u64 g1 = (u64)(uintptr_t)(gsrc[m] + (size_t)r1i * DDIM + k0 + c1i);
            ASYNC_LD_B128(l0, g0);
            ASYNC_LD_B128(l1, g1);
        }
    };

    v8f acc[8] = {};
    issue(0, 0);   // prefetch first k-tile

    for (int kt = 0; kt < 16; ++kt) {
        const int cur = kt & 1;
        if (kt < 15) {
            issue(cur ^ 1, (kt + 1) * 32);  // prefetch next tile into other buffer
            WAIT_ASYNC_8();                 // 8 newest outstanding => tile kt landed
        } else {
            WAIT_ASYNC_0();                 // last tile: drain
        }
        __syncthreads();                    // everyone's DMAs for tile kt visible

        const u16* sAh = lds[cur][0];
        const u16* sAl = lds[cur][1];
        const u16* sBh = lds[cur][2];
        const u16* sBl = lds[cur][3];

        // A fragment (16x32): lane holds row = lane%16, K-half = (lane/16)*8 and +16
        const int aoff = ((w << 4) + (lane & 15)) * 32 + ((lane >> 4) << 3);
        Frag ah, al;
        ah.q[0] = *(const uint4*)&sAh[aoff];
        ah.q[1] = *(const uint4*)&sAh[aoff + 16];
        al.q[0] = *(const uint4*)&sAl[aoff];
        al.q[1] = *(const uint4*)&sAl[aoff + 16];

#pragma unroll
        for (int ct = 0; ct < 8; ++ct) {
            // B fragment (32x16): lane = col%16, K-half = (lane/16)*8 and +16
            const int boff = ((ct << 4) + (lane & 15)) * 32 + ((lane >> 4) << 3);
            Frag bh, bl;
            bh.q[0] = *(const uint4*)&sBh[boff];
            bh.q[1] = *(const uint4*)&sBh[boff + 16];
            bl.q[0] = *(const uint4*)&sBl[boff];
            bl.q[1] = *(const uint4*)&sBl[boff + 16];
            // split product: hi*hi + hi*lo + lo*hi  (lo*lo below fp32 ulp)
            acc[ct] = __builtin_amdgcn_wmma_f32_16x16x32_bf16(false, ah.v, false, bh.v,
                                                              (short)0, acc[ct], false, false);
            acc[ct] = __builtin_amdgcn_wmma_f32_16x16x32_bf16(false, ah.v, false, bl.v,
                                                              (short)0, acc[ct], false, false);
            acc[ct] = __builtin_amdgcn_wmma_f32_16x16x32_bf16(false, al.v, false, bh.v,
                                                              (short)0, acc[ct], false, false);
        }
        __syncthreads();   // done reading buf[cur]; safe to overwrite next iter
    }

    // epilogue: scale, store sim, per-row min/max
    const float scale = 1.0f / (float)DDIM;
    const int rbase = row0 + (w << 4) + ((lane >> 4) << 3);  // + v
    const int cbase = col0 + (lane & 15);                    // + ct*16
    float rmin[8], rmax[8];
#pragma unroll
    for (int v = 0; v < 8; ++v) { rmin[v] = __builtin_inff(); rmax[v] = -__builtin_inff(); }
#pragma unroll
    for (int ct = 0; ct < 8; ++ct) {
#pragma unroll
        for (int v = 0; v < 8; ++v) {
            float val = acc[ct][v] * scale;
            out[(size_t)(rbase + v) * M_COLS + cbase + (ct << 4)] = val;
            rmin[v] = fminf(rmin[v], val);
            rmax[v] = fmaxf(rmax[v], val);
        }
    }
#pragma unroll
    for (int v = 0; v < 8; ++v) {
        float mn = rmin[v], mx = rmax[v];
#pragma unroll
        for (int m = 1; m < 16; m <<= 1) {
            mn = fminf(mn, __shfl_xor(mn, m, 32));
            mx = fmaxf(mx, __shfl_xor(mx, m, 32));
        }
        if ((lane & 15) == 0) {   // lanes 0 and 16 hold rows v and v+8
            atomicMin(&minK[rbase + v], fkey(mn));
            atomicMax(&maxK[rbase + v], fkey(mx));
        }
    }
}

// ---------- kernel 4: pointwise soft mask (float4 over d_out) ----------
__global__ __launch_bounds__(256)
void finalize_kernel(float* __restrict__ out,
                     const float* __restrict__ rowA, const float* __restrict__ rowB,
                     const float* __restrict__ rowIS,
                     const u32* __restrict__ minK, const u32* __restrict__ maxK) {
    size_t idx = (size_t)blockIdx.x * blockDim.x + threadIdx.x;  // float4 index
    int row = (int)(idx >> 11);                                   // 2048 float4 per row
    float4* p = (float4*)out + idx;
    float4 sv = *p;

    float smin = fdec(minK[row]);
    float smax = fdec(maxK[row]);
    float inv  = 1.0f / (smax - smin);
    // top_sims = max(sim_norm) = (smax-smin)/(smax-smin) = exactly 1.0f
    float a  = fminf(rowA[row], 1.0f) - EPSF;
    float b  = rowB[row];
    float is = rowIS[row];

    float r[4] = { sv.x, sv.y, sv.z, sv.w };
#pragma unroll
    for (int e = 0; e < 4; ++e) {
        float sim = r[e];
        float sn  = (sim - smin) * inv;
        float num = fmaxf(powf(fabsf(sn - a), is), EPSF);
        float den = num + fmaxf(powf(fabsf(b - sn), is), EPSF);
        float soft = num / den;
        bool lo = sn < a;
        bool hi = sn > b;
        float lm = lo ? -__builtin_inff() : logf(hi ? 1.0f : soft);
        r[e] = sim + lm;
    }
    *p = make_float4(r[0], r[1], r[2], r[3]);
}

// ---------- launch ----------
extern "C" void kernel_launch(void* const* d_in, const int* in_sizes, int n_in,
                              void* d_out, int out_size, void* d_ws, size_t ws_size,
                              hipStream_t stream) {
    const float* x    = (const float*)d_in[0];   // 4096x512
    const float* xn   = (const float*)d_in[1];   // 8192x512
    const float* W    = (const float*)d_in[2];   // 3x512
    const float* bias = (const float*)d_in[3];   // 3
    float* out = (float*)d_out;                  // 4096x8192

    // workspace carve-up (~24.1 MiB)
    u16* xhi = (u16*)d_ws;
    u16* xlo = xhi + (size_t)N_ROWS * DDIM;
    u16* nhi = xlo + (size_t)N_ROWS * DDIM;
    u16* nlo = nhi + (size_t)M_COLS * DDIM;
    float* rowA  = (float*)(nlo + (size_t)M_COLS * DDIM);
    float* rowB  = rowA + N_ROWS;
    float* rowIS = rowB + N_ROWS;
    u32* minK = (u32*)(rowIS + N_ROWS);
    u32* maxK = minK + N_ROWS;

    split_kernel<<<N_ROWS + M_COLS, 128, 0, stream>>>(x, xn, xhi, xlo, nhi, nlo);
    rowp_kernel<<<N_ROWS / 256, 256, 0, stream>>>(x, W, bias, rowA, rowB, rowIS, minK, maxK);

    dim3 grid(M_COLS / 128, N_ROWS / 128);
    gemm_wmma<<<grid, 256, 0, stream>>>(xhi, xlo, nhi, nlo, out, minK, maxK);

    finalize_kernel<<<(N_ROWS * (M_COLS / 4)) / 256, 256, 0, stream>>>(
        out, rowA, rowB, rowIS, minK, maxK);
}
